// eye_73057393705412
// MI455X (gfx1250) — compile-verified, matrix operand order checked by reading, not run on hardware
//
#include <hip/hip_runtime.h>
#include <hip/hip_bf16.h>
#include <math.h>

typedef __bf16 bf16;
typedef __attribute__((ext_vector_type(16))) __bf16 v16bf;
typedef __attribute__((ext_vector_type(8)))  float  v8f;
typedef __attribute__((ext_vector_type(4)))  unsigned int u32x4;
typedef __attribute__((ext_vector_type(8)))  int i32x8;
typedef __attribute__((ext_vector_type(4)))  int i32x4;

#define B_   64
#define S_   512
#define D_   768
#define F_   4
#define H1_  256
#define H2_  128
#define DIN1 772          // D + F
#define KP1  800          // DIN1 padded to multiple of 32
#define NG1  1024         // 4*H1
#define NG2  512          // 4*H2
#define SEQD 512          // 2*H1
#define M_   (B_*S_)      // 32768
#define BSTR 34           // padded row stride (bf16 elems) of transposed B panel in LDS

__device__ __forceinline__ float sigmoidf_(float x) { return 1.0f / (1.0f + __expf(-x)); }

// ---- WMMA fragment loaders (bf16, 16x16x32, wave32; ISA 7.12.2 layouts) ----
__device__ __forceinline__ v16bf load_frag_a(const bf16* p, int ld) {
  const int lane = threadIdx.x & 31;
  const int m  = lane & 15;
  const int kb = (lane >> 4) << 3;
  const bf16* q = p + (size_t)m * ld + kb;
  v16bf r;
#pragma unroll
  for (int i = 0; i < 8; ++i) {
    r[i]     = q[i];
    r[i + 8] = q[16 + i];
  }
  return r;
}
// B operand stored TRANSPOSED (p[n*ld + k]) -> two contiguous 16B runs per lane.
__device__ __forceinline__ v16bf load_frag_bt(const bf16* p, int ld) {
  const int lane = threadIdx.x & 31;
  const int n  = lane & 15;
  const int kb = (lane >> 4) << 3;
  const bf16* q = p + (size_t)n * ld + kb;
  v16bf r;
#pragma unroll
  for (int i = 0; i < 8; ++i) {
    r[i]     = q[i];
    r[i + 8] = q[16 + i];
  }
  return r;
}

// ---- TDM: issue a 2D bf16 tile load global->LDS (ISA ch.8 D# layout) ----
// tile_d0 x tile_d1 elements (2B each), contiguous dim stride = row_stride elems.
// This toolchain's builtin takes 6 args (g0..g4 + cpol); g2..g4 unused for 2D tiles.
__device__ __forceinline__ void tdm_load_tile_2d(unsigned int lds_byte,
                                                 unsigned long long gaddr,
                                                 unsigned int tile_d0, unsigned int tile_d1,
                                                 unsigned long long row_stride) {
  u32x4 g0;
  g0[0] = 1u;                                              // count=1 (valid user D#)
  g0[1] = lds_byte;                                        // lds_addr
  g0[2] = (unsigned int)(gaddr & 0xFFFFFFFFu);             // global_addr[31:0]
  g0[3] = (unsigned int)((gaddr >> 32) & 0x1FFFFFFu) | (2u << 30);  // addr[56:32] | type=2
  i32x8 g1;
  g1[0] = (int)(1u << 16);                                 // workgroup_mask=0, data_size=1 (2B)
  g1[1] = (int)((tile_d0 & 0xFFFFu) << 16);                // atomic_barrier=0 | tensor_dim0 lo
  g1[2] = (int)(((tile_d0 >> 16) & 0xFFFFu) | ((tile_d1 & 0xFFFFu) << 16));
  g1[3] = (int)(((tile_d1 >> 16) & 0xFFFFu) | ((tile_d0 & 0xFFFFu) << 16)); // tile_dim0
  g1[4] = (int)(tile_d1 & 0xFFFFu);                        // tile_dim1 | tile_dim2=0
  g1[5] = (int)(row_stride & 0xFFFFFFFFu);                 // tensor_dim0_stride lo32
  g1[6] = (int)((row_stride >> 32) & 0xFFFFu);             // stride hi16 | dim1_stride lo=0
  g1[7] = 0;
  i32x4 g2 = {0, 0, 0, 0};
  i32x4 g3 = {0, 0, 0, 0};
  i32x8 g4 = {0, 0, 0, 0, 0, 0, 0, 0};
  __builtin_amdgcn_tensor_load_to_lds(g0, g1, g2, g3, g4, 0);
}

// ---- f32 -> bf16 weight convert with K-padding (row-major out) ----
__global__ void convert_pad_bf16(const float* __restrict__ src, bf16* __restrict__ dst,
                                 int K, int N, int Kp) {
  size_t i = (size_t)blockIdx.x * 256 + threadIdx.x;
  size_t tot = (size_t)Kp * N;
  if (i >= tot) return;
  int k = (int)(i / N);
  int n = (int)(i % N);
  dst[i] = (k < K) ? (bf16)src[(size_t)k * N + n] : (bf16)0.0f;
}

// ---- f32 [K,N] -> bf16 transposed [N,K] (recurrent weights / WMMA B operand) ----
__global__ void convert_transpose_bf16(const float* __restrict__ src, bf16* __restrict__ dst,
                                       int K, int N) {
  size_t i = (size_t)blockIdx.x * 256 + threadIdx.x;
  size_t tot = (size_t)K * N;
  if (i >= tot) return;
  int k = (int)(i / N);
  int n = (int)(i % N);
  dst[(size_t)n * K + k] = (bf16)src[i];
}

// ---- segment-mean + concat + cast, writing X bf16 [B*S, KP1] zero padded ----
__global__ __launch_bounds__(256) void segmean_concat_kernel(
    const float* __restrict__ hidden, const float* __restrict__ lstm_input,
    const int* __restrict__ mask, bf16* __restrict__ X) {
  const int seg = blockIdx.x;
  const int b   = blockIdx.y;
  const int* mrow = &mask[b * S_];
  int lo = 0, hi = S_;
  while (lo < hi) { int mid = (lo + hi) >> 1; if (mrow[mid] < seg) lo = mid + 1; else hi = mid; }
  const int start = lo;
  hi = S_;
  while (lo < hi) { int mid = (lo + hi) >> 1; if (mrow[mid] <= seg) lo = mid + 1; else hi = mid; }
  const int end = lo;
  const int cnt = end - start;
  const float inv = cnt > 0 ? 1.0f / (float)cnt : 0.0f;
  bf16* xrow = &X[((size_t)b * S_ + seg) * KP1];
  for (int d = threadIdx.x; d < KP1; d += 256) {
    float val;
    if (d < D_) {
      float s = 0.0f;
      for (int t = start; t < end; ++t) s += hidden[((size_t)b * S_ + t) * D_ + d];
      val = s * inv;
    } else if (d < DIN1) {
      val = lstm_input[((size_t)b * S_ + seg) * F_ + (d - D_)];
    } else {
      val = 0.0f;
    }
    xrow[d] = (bf16)val;
  }
}

// ---- WMMA GEMM: C(bf16)[M,N] = A(bf16)[M,K] @ B(bf16)[K,N] + bias(f32) ----
// A panel: GLOBAL_LOAD_ASYNC_TO_LDS_B128 (ASYNCcnt).
// B panel: TENSOR_LOAD_TO_LDS (TENSORcnt) into row-major staging, then LDS->LDS transpose.
__global__ __launch_bounds__(256) void wmma_gemm_bias_bf16(
    const bf16* __restrict__ A, const bf16* __restrict__ Bm,
    const float* __restrict__ bias, bf16* __restrict__ C,
    int M, int N, int K) {
  __shared__ bf16 As[32 * 32];        // row-major A panel
  __shared__ bf16 Bs_row[32 * 64];    // row-major B panel (TDM destination)
  __shared__ bf16 Bs_t[64 * BSTR];    // transposed B panel [n][k]
  const int tid  = threadIdx.x;
  const int wave = tid >> 5, lane = tid & 31;
  const int wy = wave >> 2, wx = wave & 3;
  const int gm0 = blockIdx.y * 32, gn0 = blockIdx.x * 64;
  v8f acc = {};
  for (int k0 = 0; k0 < K; k0 += 32) {
    __syncthreads();
    // B panel 32x64 via Tensor Data Mover: one instruction per panel (wave 0)
    if (wave == 0) {
      tdm_load_tile_2d((unsigned int)(uintptr_t)(&Bs_row[0]),
                       (unsigned long long)(uintptr_t)(&Bm[(size_t)k0 * N + gn0]),
                       /*tile_d0=*/64u, /*tile_d1=*/32u, /*row_stride=*/(unsigned long long)N);
    }
    // A panel 32x32 via async global->LDS DMA, 16B per lane (threads 0..127)
    if (tid < 128) {
      int idx = tid * 8, r = idx >> 5, c = idx & 31;
      unsigned int lds_off = (unsigned int)(uintptr_t)(&As[r * 32 + c]);
      unsigned long long gaddr =
          (unsigned long long)(uintptr_t)(&A[(size_t)(gm0 + r) * K + k0 + c]);
      asm volatile("global_load_async_to_lds_b128 %0, %1, off"
                   :: "v"(lds_off), "v"(gaddr) : "memory");
    }
    if (k0 + 32 < K) {  // pull next panels toward cache (global_prefetch_b8)
      __builtin_prefetch(&A[(size_t)(gm0 + (tid >> 3)) * K + k0 + 32], 0, 0);
      __builtin_prefetch(&Bm[(size_t)(k0 + 32 + (tid >> 3)) * N + gn0], 0, 0);
    }
    asm volatile("s_wait_asynccnt 0x0" ::: "memory");
    __builtin_amdgcn_s_wait_tensorcnt(0);
    __syncthreads();                      // As + Bs_row visible to all waves
    {  // LDS->LDS transpose of the B panel (vector loads, strided b16 stores)
      int idx = tid * 8, r = idx >> 6, c = idx & 63;
      const bf16* src = &Bs_row[r * 64 + c];
#pragma unroll
      for (int j = 0; j < 8; ++j) Bs_t[(c + j) * BSTR + r] = src[j];
    }
    __syncthreads();                      // Bs_t visible
    v16bf af  = load_frag_a(&As[(wy * 16) * 32], 32);
    v16bf bfr = load_frag_bt(&Bs_t[(wx * 16) * BSTR], BSTR);
    acc = __builtin_amdgcn_wmma_f32_16x16x32_bf16(false, af, false, bfr, (short)0, acc,
                                                  false, false);
  }
  const int n = gn0 + wx * 16 + (lane & 15);
  const float bv = bias ? bias[n] : 0.0f;
  const int rb = gm0 + wy * 16 + ((lane >> 4) << 3);
#pragma unroll
  for (int v = 0; v < 8; ++v) C[(size_t)(rb + v) * N + n] = (bf16)(acc[v] + bv);
}

// ---- persistent bidirectional LSTM recurrence ----
// grid = 2 blocks (dir 0 fwd / 1 bwd), 1024 threads = 32 waves.
// h (bf16 [64,H]) + z (f32 [64,4H]) in LDS (288 KB for H=256: needs CDNA5's 320 KB/WGP).
__global__ __launch_bounds__(1024) void lstm_recurrence(
    const bf16* __restrict__ XW_f, const bf16* __restrict__ XW_b,   // [B,S,4H] bf16
    const bf16* __restrict__ WrT_f, const bf16* __restrict__ WrT_b, // [4H,H] bf16
    float* __restrict__ c_state,                                    // [2,B,H]
    bf16* __restrict__ seq_out, int seq_ld,                         // [B,S,seq_ld] or null
    float* __restrict__ hlast, int hl_ld,                           // [B,hl_ld] or null
    int H) {
  extern __shared__ char smem[];
  bf16*  h_lds = (bf16*)smem;                          // [B_,H]
  float* z_lds = (float*)(smem + (size_t)B_ * H * 2);  // [B_,4H]
  const int dir = blockIdx.x;
  const bf16* XW  = dir ? XW_b  : XW_f;
  const bf16* WrT = dir ? WrT_b : WrT_f;
  float* cst = &c_state[(size_t)dir * B_ * H];
  const int NH4 = 4 * H;
  const int tid = threadIdx.x;
  const int wave = tid >> 5, lane = tid & 31;
  const int ntn = NH4 >> 4;
  const int ntiles = 4 * ntn;
  const int kt_cnt = H >> 5;

  for (int i = tid; i < B_ * H; i += 1024) { h_lds[i] = (bf16)0.0f; cst[i] = 0.0f; }
  __syncthreads();
  // cluster-scope barrier: NOP when not launched as a cluster (ClusterID==0, ISA 3.2)
  __builtin_amdgcn_s_cluster_barrier();

  for (int t = 0; t < S_; ++t) {
    const int s = dir ? (S_ - 1 - t) : t;
    const int s_next = dir ? (s > 0 ? s - 1 : s) : (s < S_ - 1 ? s + 1 : s);
    // phase 1: z = h @ Wr + XW[:, s, :]
    for (int tix = wave; tix < ntiles; tix += 32) {
      const int mt = tix & 3;
      const int nt = tix >> 2;
      v8f acc = {};
      for (int kt = 0; kt < kt_cnt; ++kt) {
        v16bf af  = load_frag_a(&h_lds[(mt * 16) * H + kt * 32], H);
        v16bf bfr = load_frag_bt(&WrT[(size_t)(nt * 16) * H + kt * 32], H);
        acc = __builtin_amdgcn_wmma_f32_16x16x32_bf16(false, af, false, bfr, (short)0, acc,
                                                      false, false);
      }
      const int n  = nt * 16 + (lane & 15);
      const int rb = mt * 16 + ((lane >> 4) << 3);
      __builtin_prefetch(&XW[((size_t)rb * S_ + s_next) * NH4 + n], 0, 0);
#pragma unroll
      for (int v = 0; v < 8; ++v) {
        const int r = rb + v;  // batch row
        z_lds[r * NH4 + n] = acc[v] + (float)XW[((size_t)r * S_ + s) * NH4 + n];
      }
    }
    __syncthreads();
    // phase 2: gates (Keras order i,f,g,o)
    for (int idx = tid; idx < B_ * H; idx += 1024) {
      const int bb = idx / H, j = idx - bb * H;
      const float* zr = &z_lds[bb * NH4];
      const float zi = zr[j], zf = zr[H + j], zg = zr[2 * H + j], zo = zr[3 * H + j];
      const float c = sigmoidf_(zf) * cst[idx] + sigmoidf_(zi) * tanhf(zg);
      const float h = sigmoidf_(zo) * tanhf(c);
      cst[idx] = c;
      h_lds[idx] = (bf16)h;
      if (seq_out) seq_out[((size_t)bb * S_ + s) * seq_ld + dir * H + j] = (bf16)h;
      if (hlast && t == S_ - 1) hlast[(size_t)bb * hl_ld + dir * H + j] = h;
    }
    __syncthreads();
  }
}

// ---- MLP head: [64,256] -> 128 -> 64 -> 32 -> 1 (sigmoid) ----
__global__ __launch_bounds__(128) void head_kernel(
    const float* __restrict__ x, const float* __restrict__ w1, const float* __restrict__ b1,
    const float* __restrict__ w3, const float* __restrict__ b3,
    const float* __restrict__ w5, const float* __restrict__ b5,
    const float* __restrict__ w7, const float* __restrict__ b7, float* __restrict__ out) {
  __shared__ float xr[256];
  __shared__ float a1[128];
  __shared__ float a2[64];
  __shared__ float a3[32];
  const int b = blockIdx.x, tid = threadIdx.x;
  xr[tid] = x[b * 256 + tid];
  xr[tid + 128] = x[b * 256 + 128 + tid];
  __syncthreads();
  {
    float s = b1[tid];
    for (int k = 0; k < 256; ++k) s += xr[k] * w1[k * 128 + tid];
    a1[tid] = fmaxf(s, 0.0f);
  }
  __syncthreads();
  if (tid < 64) {
    float s = b3[tid];
    for (int k = 0; k < 128; ++k) s += a1[k] * w3[k * 64 + tid];
    a2[tid] = fmaxf(s, 0.0f);
  }
  __syncthreads();
  if (tid < 32) {
    float s = b5[tid];
    for (int k = 0; k < 64; ++k) s += a2[k] * w5[k * 32 + tid];
    a3[tid] = fmaxf(s, 0.0f);
  }
  __syncthreads();
  if (tid == 0) {
    float s = b7[0];
    for (int k = 0; k < 32; ++k) s += a3[k] * w7[k];
    out[b] = 1.0f / (1.0f + __expf(-s));
  }
}

extern "C" void kernel_launch(void* const* d_in, const int* in_sizes, int n_in,
                              void* d_out, int out_size, void* d_ws, size_t ws_size,
                              hipStream_t stream) {
  (void)in_sizes; (void)n_in; (void)out_size; (void)ws_size;
  const float* hidden     = (const float*)d_in[0];
  const float* lstm_input = (const float*)d_in[1];
  const int*   mask       = (const int*)d_in[2];
  const float* l1f_k = (const float*)d_in[3];
  const float* l1f_r = (const float*)d_in[4];
  const float* l1f_b = (const float*)d_in[5];
  const float* l1b_k = (const float*)d_in[6];
  const float* l1b_r = (const float*)d_in[7];
  const float* l1b_b = (const float*)d_in[8];
  const float* l2f_k = (const float*)d_in[9];
  const float* l2f_r = (const float*)d_in[10];
  const float* l2f_b = (const float*)d_in[11];
  const float* l2b_k = (const float*)d_in[12];
  const float* l2b_r = (const float*)d_in[13];
  const float* l2b_b = (const float*)d_in[14];
  const float* w1 = (const float*)d_in[15];
  const float* b1 = (const float*)d_in[16];
  const float* w3 = (const float*)d_in[17];
  const float* b3 = (const float*)d_in[18];
  const float* w5 = (const float*)d_in[19];
  const float* b5 = (const float*)d_in[20];
  const float* w7 = (const float*)d_in[21];
  const float* b7 = (const float*)d_in[22];

  char* p = (char*)d_ws;
  auto alloc = [&](size_t bytes) { char* r = p; p += (bytes + 255) & ~(size_t)255; return r; };
  bf16*  Xbf   = (bf16*)alloc((size_t)M_ * KP1 * 2);
  bf16*  Wk1f  = (bf16*)alloc((size_t)KP1 * NG1 * 2);
  bf16*  Wk1b  = (bf16*)alloc((size_t)KP1 * NG1 * 2);
  bf16*  Wk2f  = (bf16*)alloc((size_t)SEQD * NG2 * 2);
  bf16*  Wk2b  = (bf16*)alloc((size_t)SEQD * NG2 * 2);
  bf16*  WrT1f = (bf16*)alloc((size_t)H1_ * NG1 * 2);
  bf16*  WrT1b = (bf16*)alloc((size_t)H1_ * NG1 * 2);
  bf16*  WrT2f = (bf16*)alloc((size_t)H2_ * NG2 * 2);
  bf16*  WrT2b = (bf16*)alloc((size_t)H2_ * NG2 * 2);
  bf16*  XW1f  = (bf16*)alloc((size_t)M_ * NG1 * 2);
  bf16*  XW1b  = (bf16*)alloc((size_t)M_ * NG1 * 2);
  bf16*  seq   = (bf16*)alloc((size_t)M_ * SEQD * 2);
  bf16*  XW2f  = (bf16*)alloc((size_t)M_ * NG2 * 2);
  bf16*  XW2b  = (bf16*)alloc((size_t)M_ * NG2 * 2);
  float* cst     = (float*)alloc((size_t)2 * B_ * H1_ * 4);
  float* head_in = (float*)alloc((size_t)B_ * 256 * 4);

  auto conv = [&](const float* src, bf16* dst, int K, int N, int Kp) {
    size_t tot = (size_t)Kp * N;
    convert_pad_bf16<<<dim3((unsigned)((tot + 255) / 256)), dim3(256), 0, stream>>>(src, dst, K, N, Kp);
  };
  auto convT = [&](const float* src, bf16* dst, int K, int N) {
    size_t tot = (size_t)K * N;
    convert_transpose_bf16<<<dim3((unsigned)((tot + 255) / 256)), dim3(256), 0, stream>>>(src, dst, K, N);
  };
  conv(l1f_k, Wk1f, DIN1, NG1, KP1);
  conv(l1b_k, Wk1b, DIN1, NG1, KP1);
  conv(l2f_k, Wk2f, SEQD, NG2, SEQD);
  conv(l2b_k, Wk2b, SEQD, NG2, SEQD);
  convT(l1f_r, WrT1f, H1_, NG1);
  convT(l1b_r, WrT1b, H1_, NG1);
  convT(l2f_r, WrT2f, H2_, NG2);
  convT(l2b_r, WrT2b, H2_, NG2);

  segmean_concat_kernel<<<dim3(S_, B_), dim3(256), 0, stream>>>(hidden, lstm_input, mask, Xbf);

  wmma_gemm_bias_bf16<<<dim3(NG1 / 64, M_ / 32), dim3(256), 0, stream>>>(Xbf, Wk1f, l1f_b, XW1f, M_, NG1, KP1);
  wmma_gemm_bias_bf16<<<dim3(NG1 / 64, M_ / 32), dim3(256), 0, stream>>>(Xbf, Wk1b, l1b_b, XW1b, M_, NG1, KP1);

  size_t shm1 = (size_t)B_ * H1_ * 2 + (size_t)B_ * 4 * H1_ * 4;
  (void)hipFuncSetAttribute((const void*)lstm_recurrence,
                            hipFuncAttributeMaxDynamicSharedMemorySize, (int)shm1);
  lstm_recurrence<<<dim3(2), dim3(1024), shm1, stream>>>(XW1f, XW1b, WrT1f, WrT1b, cst,
                                                         seq, SEQD, nullptr, 0, H1_);

  wmma_gemm_bias_bf16<<<dim3(NG2 / 64, M_ / 32), dim3(256), 0, stream>>>(seq, Wk2f, l2f_b, XW2f, M_, NG2, SEQD);
  wmma_gemm_bias_bf16<<<dim3(NG2 / 64, M_ / 32), dim3(256), 0, stream>>>(seq, Wk2b, l2b_b, XW2b, M_, NG2, SEQD);

  size_t shm2 = (size_t)B_ * H2_ * 2 + (size_t)B_ * 4 * H2_ * 4;
  lstm_recurrence<<<dim3(2), dim3(1024), shm2, stream>>>(XW2f, XW2b, WrT2f, WrT2b, cst,
                                                         nullptr, 0, head_in, 256, H2_);

  head_kernel<<<dim3(B_), dim3(128), 0, stream>>>(head_in, w1, b1, w3, b3, w5, b5, w7, b7,
                                                  (float*)d_out);
}
